// CharRNN_19653770346501
// MI455X (gfx1250) — matrix operand order repaired and checked by reading, not compiled, not run on hardware
//
#include <hip/hip_runtime.h>
#include <hip/hip_bf16.h>
#include <math.h>

#define VSZ  16384
#define EDIM 256
#define HDIM 1024
#define LNUM 2
#define BB   128
#define TT   256
#define SS   64
#define BT   (BB * TT)

typedef __attribute__((ext_vector_type(16))) __bf16 v16bf;
typedef __attribute__((ext_vector_type(8)))  float  v8f;

union FragU { v16bf v; uint4 q[2]; };

__device__ __forceinline__ unsigned short f2bf(float f) {
  unsigned int u = __float_as_uint(f);
  u += 0x7FFFu + ((u >> 16) & 1u);        // round-to-nearest-even
  return (unsigned short)(u >> 16);
}
__device__ __forceinline__ float bf2f(unsigned short s) {
  return __uint_as_float(((unsigned int)s) << 16);
}
__device__ __forceinline__ float softplusf(float x) {
  return fmaxf(x, 0.0f) + log1pf(expf(-fabsf(x)));
}
__device__ __forceinline__ float sigmoidf_(float x) {
  return 1.0f / (1.0f + expf(-x));
}

// WGP-scope prefetch (SCOPE=0): pulls into all cache levels incl. WGP$.
// __builtin_prefetch can only produce SE/SYS scope, which skips the WGP caches,
// so emit the ISA instruction directly (prefetch does not touch LOADcnt).
__device__ __forceinline__ void prefetch_wgp(const void* p) {
  asm volatile("global_prefetch_b8 %0, off" :: "v"(p));
}

__device__ __forceinline__ v8f wmma_bf16(const FragU& a, const FragU& b, v8f c) {
  return __builtin_amdgcn_wmma_f32_16x16x32_bf16(false, a.v, false, b.v,
                                                 (short)0, c, false, false);
}

// ---------------------------------------------------------------------------
// 16(M) x 64(N) strip: used by the small latency-critical kernels (recurrence,
// NCE). A: bf16 [M,K] row-major; Bt: bf16 [N,K] row-major (B transposed).
// Fragment layouts per CDNA5 ISA 7.12.2 (wave32).
// ---------------------------------------------------------------------------
__device__ __forceinline__ void wmma_strip16x64(const unsigned short* __restrict__ A,
                                                const unsigned short* __restrict__ Bt,
                                                int K, int row0, int col0, v8f acc[4]) {
  const int lane = threadIdx.x & 31;
  const int half = lane >> 4;
  const int mr   = lane & 15;
  const unsigned short* arow = A  + (size_t)(row0 + mr) * K;
  const unsigned short* brow = Bt + (size_t)(col0 + mr) * K;
  for (int k0 = 0; k0 < K; k0 += 32) {
    FragU a, b[4];
    a.q[0] = *(const uint4*)(arow + k0 + half * 8);
    a.q[1] = *(const uint4*)(arow + k0 + 16 + half * 8);
#pragma unroll
    for (int tj = 0; tj < 4; ++tj) {
      const unsigned short* bp = brow + (size_t)tj * 16 * K + k0;
      b[tj].q[0] = *(const uint4*)(bp + half * 8);
      b[tj].q[1] = *(const uint4*)(bp + 16 + half * 8);
    }
#pragma unroll
    for (int tj = 0; tj < 4; ++tj) acc[tj] = wmma_bf16(a, b[tj], acc[tj]);
  }
}

// ---------------------------------------------------------------------------
// 32(M) x 64(N) strip: higher arithmetic intensity for the big [32768,*] GEMMs.
// 2 A-fragments reuse the same 4 B-fragments -> 8 WMMA per 12 b128 loads.
// ---------------------------------------------------------------------------
__device__ __forceinline__ void wmma_strip32x64(const unsigned short* __restrict__ A,
                                                const unsigned short* __restrict__ Bt,
                                                int K, int row0, int col0, v8f acc[8]) {
  const int lane = threadIdx.x & 31;
  const int half = lane >> 4;
  const int mr   = lane & 15;
  const unsigned short* arow0 = A + (size_t)(row0 + mr) * K;
  const unsigned short* arow1 = A + (size_t)(row0 + 16 + mr) * K;
  const unsigned short* brow  = Bt + (size_t)(col0 + mr) * K;
  for (int k0 = 0; k0 < K; k0 += 32) {
    // WGP-scope prefetch one k-step ahead: data is L2-resident, so this
    // stages the streams into the WGP$ before the fragment loads hit them.
    if (k0 + 32 < K) {
      prefetch_wgp(arow0 + k0 + 32);
      prefetch_wgp(arow1 + k0 + 32);
      prefetch_wgp(brow + k0 + 32);
    }
    FragU a0, a1, b[4];
    a0.q[0] = *(const uint4*)(arow0 + k0 + half * 8);
    a0.q[1] = *(const uint4*)(arow0 + k0 + 16 + half * 8);
    a1.q[0] = *(const uint4*)(arow1 + k0 + half * 8);
    a1.q[1] = *(const uint4*)(arow1 + k0 + 16 + half * 8);
#pragma unroll
    for (int tj = 0; tj < 4; ++tj) {
      const unsigned short* bp = brow + (size_t)tj * 16 * K + k0;
      b[tj].q[0] = *(const uint4*)(bp + half * 8);
      b[tj].q[1] = *(const uint4*)(bp + 16 + half * 8);
    }
#pragma unroll
    for (int tj = 0; tj < 4; ++tj) {
      acc[tj]     = wmma_bf16(a0, b[tj], acc[tj]);
      acc[4 + tj] = wmma_bf16(a1, b[tj], acc[4 + tj]);
    }
  }
}

// ---------------------------------------------------------------------------
// Utility kernels
// ---------------------------------------------------------------------------
__global__ void k_zero_scalar(float* p) { p[0] = 0.0f; }

__global__ void k_zero_h(float* hf, unsigned short* hbf) {
  int i = blockIdx.x * blockDim.x + threadIdx.x;
  if (i < BB * HDIM) { hf[i] = 0.0f; hbf[i] = 0; }
}

// src f32 [R][C] -> dst bf16 [C][R]  (transpose + convert)
__global__ void k_cvtT(const float* __restrict__ src, unsigned short* __restrict__ dst,
                       int R, int C) {
  int i = blockIdx.x * blockDim.x + threadIdx.x;
  if (i < R * C) {
    int r = i / C, c = i - r * C;
    dst[(size_t)c * R + r] = f2bf(src[i]);
  }
}

// Xe[i][e] = bf16(embedding[input[i]][e]),  i = b*T+t
__global__ void k_embed(const int* __restrict__ idx, const float* __restrict__ emb,
                        unsigned short* __restrict__ Xe) {
  int i = blockIdx.x * blockDim.x + threadIdx.x;
  if (i < BT * EDIM) {
    int row = i / EDIM, e = i - row * EDIM;
    Xe[i] = f2bf(emb[(size_t)idx[row] * EDIM + e]);
  }
}

// SampBt[s][k] = bf16(softmax_w[nce[s]][k])   (already the [N][K] layout we need)
__global__ void k_samp_gather(const int* __restrict__ nce, const float* __restrict__ sw,
                              unsigned short* __restrict__ SampBt) {
  int i = blockIdx.x * blockDim.x + threadIdx.x;
  if (i < SS * HDIM) {
    int s = i / HDIM, k = i - s * HDIM;
    SampBt[i] = f2bf(sw[(size_t)nce[s] * HDIM + k]);
  }
}

// ---------------------------------------------------------------------------
// Big bf16 WMMA GEMM: C[M,N] = A[M,K] * Bt[N,K]^T (+bias). 1 wave / block,
// 32x64 tile. grid = (N/64, M/32), block = 32
// ---------------------------------------------------------------------------
__global__ void k_gemm(const unsigned short* __restrict__ A,
                       const unsigned short* __restrict__ Bt,
                       int N, int K, const float* __restrict__ bias,
                       float* __restrict__ Cf, unsigned short* __restrict__ Cb) {
  const int row0 = blockIdx.y * 32;
  const int col0 = blockIdx.x * 64;
  const int lane = threadIdx.x & 31;
  const int half = lane >> 4;
  const int mr   = lane & 15;
  v8f acc[8];
  const v8f z = {0.f, 0.f, 0.f, 0.f, 0.f, 0.f, 0.f, 0.f};
#pragma unroll
  for (int i = 0; i < 8; ++i) acc[i] = z;
  wmma_strip32x64(A, Bt, K, row0, col0, acc);
#pragma unroll
  for (int mi = 0; mi < 2; ++mi) {
#pragma unroll
    for (int tj = 0; tj < 4; ++tj) {
#pragma unroll
      for (int v = 0; v < 8; ++v) {
        int r = row0 + mi * 16 + v + 8 * half;
        int c = col0 + tj * 16 + mr;
        float val = acc[mi * 4 + tj][v];
        if (bias) val += bias[c];
        if (Cf) Cf[(size_t)r * N + c] = val;
        if (Cb) Cb[(size_t)r * N + c] = f2bf(val);
      }
    }
  }
}

// ---------------------------------------------------------------------------
// Recurrent gate step: gh = h @ Whg (fused MI-sigmoid epilogue)
// grid = (2H/64, B/16), block = 32
// ---------------------------------------------------------------------------
__global__ void k_gates(const unsigned short* __restrict__ hbf,
                        const unsigned short* __restrict__ WhgT,
                        const float* __restrict__ Gx, const float* __restrict__ hf,
                        const float* __restrict__ ag, const float* __restrict__ b1g,
                        const float* __restrict__ b2g, const float* __restrict__ bg,
                        unsigned short* __restrict__ rhbf, float* __restrict__ uf, int t) {
  const int row0 = blockIdx.y * 16;
  const int col0 = blockIdx.x * 64;
  const int lane = threadIdx.x & 31;
  const int half = lane >> 4;
  const int mr   = lane & 15;
  v8f acc[4];
  const v8f z = {0.f, 0.f, 0.f, 0.f, 0.f, 0.f, 0.f, 0.f};
#pragma unroll
  for (int i = 0; i < 4; ++i) acc[i] = z;
  wmma_strip16x64(hbf, WhgT, HDIM, row0, col0, acc);
#pragma unroll
  for (int tj = 0; tj < 4; ++tj) {
#pragma unroll
    for (int v = 0; v < 8; ++v) {
      int b = row0 + v + 8 * half;
      int c = col0 + tj * 16 + mr;
      float gh = acc[tj][v];
      float gx = Gx[((size_t)b * TT + t) * (2 * HDIM) + c];
      float g  = sigmoidf_(ag[c] * gx * gh + b1g[c] * gx + b2g[c] * gh + bg[c]);
      if (c < HDIM) rhbf[b * HDIM + c] = f2bf(g * hf[b * HDIM + c]);   // reset gate
      else          uf[b * HDIM + (c - HDIM)] = g;                     // update gate
    }
  }
}

// ---------------------------------------------------------------------------
// Candidate + state update: ch = (r*h) @ Whc, c = tanh(MI), h = u*h+(1-u)*c
// grid = (H/64, B/16), block = 32
// ---------------------------------------------------------------------------
__global__ void k_cand(const unsigned short* __restrict__ rhbf,
                       const unsigned short* __restrict__ WhcT,
                       const float* __restrict__ Cx, const float* __restrict__ uf,
                       const float* __restrict__ ac, const float* __restrict__ b1c,
                       const float* __restrict__ b2c, const float* __restrict__ bc,
                       float* __restrict__ hf, unsigned short* __restrict__ hbf,
                       unsigned short* __restrict__ outbf, int t) {
  const int row0 = blockIdx.y * 16;
  const int col0 = blockIdx.x * 64;
  const int lane = threadIdx.x & 31;
  const int half = lane >> 4;
  const int mr   = lane & 15;
  v8f acc[4];
  const v8f z = {0.f, 0.f, 0.f, 0.f, 0.f, 0.f, 0.f, 0.f};
#pragma unroll
  for (int i = 0; i < 4; ++i) acc[i] = z;
  wmma_strip16x64(rhbf, WhcT, HDIM, row0, col0, acc);
#pragma unroll
  for (int tj = 0; tj < 4; ++tj) {
#pragma unroll
    for (int v = 0; v < 8; ++v) {
      int b = row0 + v + 8 * half;
      int c = col0 + tj * 16 + mr;
      float ch = acc[tj][v];
      float cx = Cx[((size_t)b * TT + t) * HDIM + c];
      float cc = tanhf(ac[c] * cx * ch + b1c[c] * cx + b2c[c] * ch + bc[c]);
      float u  = uf[b * HDIM + c];
      float h  = hf[b * HDIM + c];
      float hn = u * h + (1.0f - u) * cc;
      hf[b * HDIM + c] = hn;
      unsigned short hb = f2bf(hn);
      hbf[b * HDIM + c] = hb;
      outbf[((size_t)b * TT + t) * HDIM + c] = hb;
    }
  }
}

// ---------------------------------------------------------------------------
// NCE sampled logits: [BT,H] @ SampBt[S,H]^T, fused softplus + reduce.
// grid = BT/16, block = 32 (one wave -> full 16x64 strip)
// ---------------------------------------------------------------------------
__global__ void k_nce_samp(const unsigned short* __restrict__ Out1,
                           const unsigned short* __restrict__ SampBt,
                           const int* __restrict__ nce, const float* __restrict__ smb,
                           float* __restrict__ loss_acc) {
  const int row0 = blockIdx.x * 16;
  const int lane = threadIdx.x & 31;
  const int mr   = lane & 15;
  v8f acc[4];
  const v8f z = {0.f, 0.f, 0.f, 0.f, 0.f, 0.f, 0.f, 0.f};
#pragma unroll
  for (int i = 0; i < 4; ++i) acc[i] = z;
  wmma_strip16x64(Out1, SampBt, HDIM, row0, 0, acc);
  float part = 0.0f;
#pragma unroll
  for (int tj = 0; tj < 4; ++tj) {
    float sb = smb[nce[tj * 16 + mr]];
#pragma unroll
    for (int v = 0; v < 8; ++v) part += softplusf(acc[tj][v] + sb);
  }
  for (int m = 16; m >= 1; m >>= 1) part += __shfl_xor(part, m, 32);
  if (lane == 0) atomicAdd(loss_acc, part);
}

// One wave per token row: true logit dot + softplus(-logit)
__global__ void k_true_logit(const unsigned short* __restrict__ Out1,
                             const float* __restrict__ sw, const float* __restrict__ smb,
                             const int* __restrict__ tgt, float* __restrict__ loss_acc) {
  const int i = blockIdx.x;
  const int lane = threadIdx.x & 31;
  const int label = tgt[i];
  const unsigned short* orow = Out1 + (size_t)i * HDIM;
  const float* wrow = sw + (size_t)label * HDIM;
  float dot = 0.0f;
  for (int k = lane; k < HDIM; k += 32) dot += bf2f(orow[k]) * wrow[k];
  for (int m = 16; m >= 1; m >>= 1) dot += __shfl_xor(dot, m, 32);
  if (lane == 0) atomicAdd(loss_acc, softplusf(-(dot + smb[label])));
}

__global__ void k_finalize(const float* loss_acc, float* out) {
  out[0] = loss_acc[0] * (1.0f / (float)(BB * TT));
}

// ---------------------------------------------------------------------------
extern "C" void kernel_launch(void* const* d_in, const int* in_sizes, int n_in,
                              void* d_out, int out_size, void* d_ws, size_t ws_size,
                              hipStream_t stream) {
  const int*   input_data = (const int*)  d_in[0];
  const int*   targets    = (const int*)  d_in[1];
  const int*   nce        = (const int*)  d_in[2];
  const float* embedding  = (const float*)d_in[3];
  const float* win        = (const float*)d_in[4];
  const float* bin_       = (const float*)d_in[5];
  const float* Wxg        = (const float*)d_in[6];
  const float* Whg        = (const float*)d_in[7];
  const float* ag         = (const float*)d_in[8];
  const float* b1g        = (const float*)d_in[9];
  const float* b2g        = (const float*)d_in[10];
  const float* bg         = (const float*)d_in[11];
  const float* Wxc        = (const float*)d_in[12];
  const float* Whc        = (const float*)d_in[13];
  const float* ac         = (const float*)d_in[14];
  const float* b1c        = (const float*)d_in[15];
  const float* b2c        = (const float*)d_in[16];
  const float* bc         = (const float*)d_in[17];
  const float* softmax_w  = (const float*)d_in[18];
  const float* softmax_b  = (const float*)d_in[19];

  // ---- workspace carve-up (256B aligned) ----
  char* base = (char*)d_ws;
  size_t off = 0;
  auto alloc = [&](size_t bytes) -> void* {
    void* p = base + off;
    off = (off + bytes + 255) & ~(size_t)255;
    return p;
  };
  unsigned short* WinT    = (unsigned short*)alloc((size_t)HDIM * EDIM * 2);
  unsigned short* WxgT[LNUM], *WhgT[LNUM], *WxcT[LNUM], *WhcT[LNUM];
  for (int l = 0; l < LNUM; ++l) {
    WxgT[l] = (unsigned short*)alloc((size_t)2 * HDIM * HDIM * 2);
    WhgT[l] = (unsigned short*)alloc((size_t)2 * HDIM * HDIM * 2);
    WxcT[l] = (unsigned short*)alloc((size_t)HDIM * HDIM * 2);
    WhcT[l] = (unsigned short*)alloc((size_t)HDIM * HDIM * 2);
  }
  unsigned short* Xe     = (unsigned short*)alloc((size_t)BT * EDIM * 2);
  unsigned short* Xbf    = (unsigned short*)alloc((size_t)BT * HDIM * 2);
  unsigned short* Out0   = (unsigned short*)alloc((size_t)BT * HDIM * 2);
  unsigned short* Out1   = (unsigned short*)alloc((size_t)BT * HDIM * 2);
  float*          Gx     = (float*)        alloc((size_t)BT * 2 * HDIM * 4);
  float*          Cx     = (float*)        alloc((size_t)BT * HDIM * 4);
  float*          hf     = (float*)        alloc((size_t)BB * HDIM * 4);
  unsigned short* hbf    = (unsigned short*)alloc((size_t)BB * HDIM * 2);
  unsigned short* rhbf   = (unsigned short*)alloc((size_t)BB * HDIM * 2);
  float*          uf     = (float*)        alloc((size_t)BB * HDIM * 4);
  unsigned short* SampBt = (unsigned short*)alloc((size_t)SS * HDIM * 2);
  float*          lossA  = (float*)        alloc(256);

  k_zero_scalar<<<1, 1, 0, stream>>>(lossA);

  // ---- weight conversion (f32 -> bf16, transposed to [N][K]) ----
  const int THR = 256;
  k_cvtT<<<(EDIM * HDIM + THR - 1) / THR, THR, 0, stream>>>(win, WinT, EDIM, HDIM);
  for (int l = 0; l < LNUM; ++l) {
    k_cvtT<<<(HDIM * 2 * HDIM + THR - 1) / THR, THR, 0, stream>>>(
        Wxg + (size_t)l * HDIM * 2 * HDIM, WxgT[l], HDIM, 2 * HDIM);
    k_cvtT<<<(HDIM * 2 * HDIM + THR - 1) / THR, THR, 0, stream>>>(
        Whg + (size_t)l * HDIM * 2 * HDIM, WhgT[l], HDIM, 2 * HDIM);
    k_cvtT<<<(HDIM * HDIM + THR - 1) / THR, THR, 0, stream>>>(
        Wxc + (size_t)l * HDIM * HDIM, WxcT[l], HDIM, HDIM);
    k_cvtT<<<(HDIM * HDIM + THR - 1) / THR, THR, 0, stream>>>(
        Whc + (size_t)l * HDIM * HDIM, WhcT[l], HDIM, HDIM);
  }

  // ---- embedding + input projection: X = emb[idx] @ win + bin ----
  k_embed<<<(BT * EDIM + THR - 1) / THR, THR, 0, stream>>>(input_data, embedding, Xe);
  k_gemm<<<dim3(HDIM / 64, BT / 32), 32, 0, stream>>>(Xe, WinT, HDIM, EDIM, bin_,
                                                      nullptr, Xbf);

  // ---- MI-GRU layers (layer-sequential; hoist gx/cx out of the time loop) ----
  for (int l = 0; l < LNUM; ++l) {
    const unsigned short* inbf  = (l == 0) ? Xbf : Out0;
    unsigned short*       outbf = (l == 0) ? Out0 : Out1;
    k_gemm<<<dim3(2 * HDIM / 64, BT / 32), 32, 0, stream>>>(inbf, WxgT[l], 2 * HDIM,
                                                            HDIM, nullptr, Gx, nullptr);
    k_gemm<<<dim3(HDIM / 64, BT / 32), 32, 0, stream>>>(inbf, WxcT[l], HDIM, HDIM,
                                                        nullptr, Cx, nullptr);
    k_zero_h<<<(BB * HDIM + THR - 1) / THR, THR, 0, stream>>>(hf, hbf);
    const float* agl = ag + l * 2 * HDIM;  const float* b1gl = b1g + l * 2 * HDIM;
    const float* b2gl = b2g + l * 2 * HDIM; const float* bgl = bg + l * 2 * HDIM;
    const float* acl = ac + l * HDIM;      const float* b1cl = b1c + l * HDIM;
    const float* b2cl = b2c + l * HDIM;    const float* bcl = bc + l * HDIM;
    for (int t = 0; t < TT; ++t) {
      k_gates<<<dim3(2 * HDIM / 64, BB / 16), 32, 0, stream>>>(
          hbf, WhgT[l], Gx, hf, agl, b1gl, b2gl, bgl, rhbf, uf, t);
      k_cand<<<dim3(HDIM / 64, BB / 16), 32, 0, stream>>>(
          rhbf, WhcT[l], Cx, uf, acl, b1cl, b2cl, bcl, hf, hbf, outbf, t);
    }
  }

  // ---- NCE loss ----
  k_samp_gather<<<(SS * HDIM + THR - 1) / THR, THR, 0, stream>>>(nce, softmax_w, SampBt);
  k_nce_samp<<<BT / 16, 32, 0, stream>>>(Out1, SampBt, nce, softmax_b, lossA);
  k_true_logit<<<BT, 32, 0, stream>>>(Out1, softmax_w, softmax_b, targets, lossA);
  k_finalize<<<1, 1, 0, stream>>>(lossA, (float*)d_out);

  (void)in_sizes; (void)n_in; (void)out_size; (void)ws_size;
}
